// UnconditionalModel_26800595927067
// MI455X (gfx1250) — matrix operand order repaired and checked by reading, compiled, not run on hardware
//
#include <hip/hip_runtime.h>
#include <hip/hip_bf16.h>

// ---------------------------------------------------------------------------
// CDNA5 (gfx1250) wave32 WMMA types
// ---------------------------------------------------------------------------
typedef __attribute__((ext_vector_type(16))) __bf16 v16bf;
typedef __attribute__((ext_vector_type(8)))  __bf16 v8bf;
typedef __attribute__((ext_vector_type(8)))  float  v8f;

union Frag {
    v8bf h[2];
    v16bf v;
};

__device__ __forceinline__ unsigned int f32_to_bf16_rne(float f) {
    unsigned int u = __builtin_bit_cast(unsigned int, f);
    unsigned int r = u + 0x7FFFu + ((u >> 16) & 1u);   // round-to-nearest-even
    return r >> 16;
}

// ---------------------------------------------------------------------------
// f32 -> bf16 row-major conversion (8 elements/thread, b128 in / b128 out)
// ---------------------------------------------------------------------------
__global__ void cvt_bf16(const float* __restrict__ x, unsigned int* __restrict__ y, int total8) {
    int i = blockIdx.x * blockDim.x + threadIdx.x;
    if (i >= total8) return;
    const float4* p = (const float4*)(x + (size_t)i * 8);
    float4 f0 = p[0];
    float4 f1 = p[1];
    uint4 o;
    o.x = f32_to_bf16_rne(f0.x) | (f32_to_bf16_rne(f0.y) << 16);
    o.y = f32_to_bf16_rne(f0.z) | (f32_to_bf16_rne(f0.w) << 16);
    o.z = f32_to_bf16_rne(f1.x) | (f32_to_bf16_rne(f1.y) << 16);
    o.w = f32_to_bf16_rne(f1.z) | (f32_to_bf16_rne(f1.w) << 16);
    *(uint4*)(y + (size_t)i * 4) = o;
}

// ---------------------------------------------------------------------------
// Pack W[K x DOUT] (f32) into per-lane WMMA B-fragment layout (bf16):
//   Wp[((t*(K/32) + kb)*32 + lane)*16 + e] = bf16( W[(kb*32 + 16*(lane>>4) + e)*DOUT
//                                                    + t*16 + (lane&15)] )
// so a B fragment is 32 contiguous bytes per lane (1KB per wave, coalesced).
// ---------------------------------------------------------------------------
template <int K, int DOUT>
__global__ void pack_w(const float* __restrict__ W, unsigned short* __restrict__ Wp) {
    constexpr int KB = K / 32;
    int idx = blockIdx.x * blockDim.x + threadIdx.x;   // over K*DOUT
    if (idx >= K * DOUT) return;
    int frag = idx >> 9;           // 512 elements per fragment
    int rem  = idx & 511;
    int lane = rem >> 4;
    int e    = rem & 15;
    int t    = frag / KB;
    int kb   = frag - t * KB;
    int k    = kb * 32 + ((lane >> 4) << 4) + e;
    int c    = t * 16 + (lane & 15);
    Wp[idx] = (unsigned short)f32_to_bf16_rne(W[(size_t)k * DOUT + c]);
}

// ---------------------------------------------------------------------------
// GEMM: Y[n x DOUT] = Xb[n x K](bf16) @ Wp(packed bf16)   f32 accumulate
// blockDim = 128 (4 waves); each wave owns one 16-row tile, all column tiles.
// A (16x32 bf16): row = lane&15; k = e + 16*(lane>>4)
//  -> two contiguous 8-element (16B) b128 loads per lane.
// ---------------------------------------------------------------------------
template <int K, int DOUT>
__global__ __launch_bounds__(128)
void gemm_bf16_wmma(const __bf16* __restrict__ Xb, const __bf16* __restrict__ Wp,
                    float* __restrict__ Y, int nrows) {
    constexpr int NT = DOUT / 16;
    constexpr int KB = K / 32;
    const int wave = threadIdx.x >> 5;
    const int lane = threadIdx.x & 31;
    const int row0 = (blockIdx.x * 4 + wave) * 16;
    if (row0 + 16 > nrows) return;         // wave-uniform: EXEC stays all-1s

    const int r    = lane & 15;
    const int half = lane >> 4;

    v8f acc[NT];
#pragma unroll
    for (int t = 0; t < NT; ++t) {
#pragma unroll
        for (int e = 0; e < 8; ++e) acc[t][e] = 0.0f;
    }

#pragma unroll
    for (int kb = 0; kb < KB; ++kb) {
        const __bf16* ap = Xb + (size_t)(row0 + r) * K + kb * 32 + half * 16;
        Frag a;
        a.h[0] = *(const v8bf*)(ap);
        a.h[1] = *(const v8bf*)(ap + 8);

#pragma unroll
        for (int t = 0; t < NT; ++t) {
            const __bf16* bp = Wp + ((size_t)(t * KB + kb) * 32 + lane) * 16;
            Frag b;
            b.h[0] = *(const v8bf*)(bp);
            b.h[1] = *(const v8bf*)(bp + 8);

            acc[t] = __builtin_amdgcn_wmma_f32_16x16x32_bf16(
                /*neg_a=*/false, a.v, /*neg_b=*/false, b.v,
                /*c_mod=*/(short)0, acc[t],
                /*reuse_a=*/false, /*reuse_b=*/false);
        }
    }

#pragma unroll
    for (int t = 0; t < NT; ++t) {
#pragma unroll
        for (int e = 0; e < 8; ++e) {
            Y[(size_t)(row0 + e + 8 * half) * DOUT + t * 16 + r] = acc[t][e];
        }
    }
}

// ---------------------------------------------------------------------------
// Graph helpers
// ---------------------------------------------------------------------------
__global__ void fill_f32(float* __restrict__ p, float v, int n) {
    int i = blockIdx.x * blockDim.x + threadIdx.x;
    if (i < n) p[i] = v;
}

__global__ void degree_kernel(const int* __restrict__ dst, float* __restrict__ deg, int E) {
    int i = blockIdx.x * blockDim.x + threadIdx.x;
    if (i < E) atomicAdd(&deg[dst[i]], 1.0f);
}

__global__ void rsqrt_kernel(float* __restrict__ p, int n) {
    int i = blockIdx.x * blockDim.x + threadIdx.x;
    if (i < n) {
        float d = p[i];
        p[i] = (d > 0.0f) ? rsqrtf(d) : 0.0f;
    }
}

// out[i,c] = H[i,c]*dinv[i]^2 + bias[c]  (self-loop + bias, initializes out)
// vectorized: 4 columns per thread, b128 in / b128 out
template <int DOUT>
__global__ void init_self_bias(const float* __restrict__ Hbuf, const float* __restrict__ dinv,
                               const float* __restrict__ bias, float* __restrict__ out, int n) {
    int idx = blockIdx.x * blockDim.x + threadIdx.x;       // over n*DOUT/4
    if (idx >= n * (DOUT / 4)) return;
    int i  = idx / (DOUT / 4);
    int c4 = idx - i * (DOUT / 4);
    float di = dinv[i];
    float d2 = di * di;
    float4 h = *(const float4*)(Hbuf + (size_t)idx * 4);
    const float4 b = *(const float4*)(bias + c4 * 4);
    float4 o;
    o.x = h.x * d2 + b.x;
    o.y = h.y * d2 + b.y;
    o.z = h.z * d2 + b.z;
    o.w = h.w * d2 + b.w;
    *(float4*)(out + (size_t)idx * 4) = o;
}

// wave-per-edge scatter: out[dst] += H[src] * dinv[src]*dinv[dst]
// lane gathers one contiguous float4/float2 (b128/b64), scatters f32 atomics
template <int DOUT>
__global__ void agg_edges(const float* __restrict__ Hbuf, const int* __restrict__ src,
                          const int* __restrict__ dst, const float* __restrict__ dinv,
                          float* __restrict__ out, int nedges) {
    int wid  = (blockIdx.x * blockDim.x + threadIdx.x) >> 5;
    int lane = threadIdx.x & 31;
    if (wid >= nedges) return;
    int s = src[wid];
    int d = dst[wid];
    float nrm = dinv[s] * dinv[d];
    if constexpr (DOUT == 128) {
        float4 h = *(const float4*)(Hbuf + (size_t)s * DOUT + lane * 4);
        float* op = out + (size_t)d * DOUT + lane * 4;
        atomicAdd(op + 0, h.x * nrm);
        atomicAdd(op + 1, h.y * nrm);
        atomicAdd(op + 2, h.z * nrm);
        atomicAdd(op + 3, h.w * nrm);
    } else {
        float2 h = *(const float2*)(Hbuf + (size_t)s * DOUT + lane * 2);
        float* op = out + (size_t)d * DOUT + lane * 2;
        atomicAdd(op + 0, h.x * nrm);
        atomicAdd(op + 1, h.y * nrm);
    }
}

// z = noise * exp(0.5*logvar) + mean
__global__ void reparam_kernel(const float* __restrict__ noise, const float* __restrict__ mean,
                               const float* __restrict__ logvar, float* __restrict__ z, int total) {
    int i = blockIdx.x * blockDim.x + threadIdx.x;
    if (i < total) z[i] = noise[i] * __expf(0.5f * logvar[i]) + mean[i];
}

// ---------------------------------------------------------------------------
// Host-side layer driver
// ---------------------------------------------------------------------------
static inline int ceil_div(long a, long b) { return (int)((a + b - 1) / b); }

template <int K, int DOUT>
static void run_gcn(const float* X, const float* W, const float* b,
                    const int* src, const int* dst, const float* dinv,
                    unsigned short* Xb, unsigned short* Wp,
                    float* tmp, float* out, int n, int E, hipStream_t stream) {
    // convert activations to bf16 row-major
    int total8 = (n * K) / 8;
    cvt_bf16<<<ceil_div(total8, 256), 256, 0, stream>>>(X, (unsigned int*)Xb, total8);
    // pack weights into B-fragment layout
    pack_w<K, DOUT><<<ceil_div(K * DOUT, 256), 256, 0, stream>>>(W, Wp);
    // WMMA GEMM
    int rowTiles = ceil_div(n, 16);
    gemm_bf16_wmma<K, DOUT><<<ceil_div(rowTiles, 4), 128, 0, stream>>>(
        (const __bf16*)Xb, (const __bf16*)Wp, tmp, n);
    // self-loop + bias init, then edge scatter
    init_self_bias<DOUT><<<ceil_div(n * (DOUT / 4), 256), 256, 0, stream>>>(tmp, dinv, b, out, n);
    agg_edges<DOUT><<<ceil_div((long)E * 32, 256), 256, 0, stream>>>(tmp, src, dst, dinv, out, E);
}

extern "C" void kernel_launch(void* const* d_in, const int* in_sizes, int n_in,
                              void* d_out, int out_size, void* d_ws, size_t ws_size,
                              hipStream_t stream) {
    (void)n_in; (void)out_size; (void)ws_size;

    const float* feature = (const float*)d_in[0];
    const int*   edge    = (const int*)d_in[1];
    const float* noise   = (const float*)d_in[2];
    const float* W_enc   = (const float*)d_in[3];
    const float* b_enc   = (const float*)d_in[4];
    const float* W_mean  = (const float*)d_in[5];
    const float* b_mean  = (const float*)d_in[6];
    const float* W_lv    = (const float*)d_in[7];
    const float* b_lv    = (const float*)d_in[8];
    const float* W_d1    = (const float*)d_in[9];
    const float* b_d1    = (const float*)d_in[10];
    const float* W_d2    = (const float*)d_in[11];
    const float* b_d2    = (const float*)d_in[12];

    const int F = 128, L = 64;
    const int n = in_sizes[0] / F;
    const int E = in_sizes[1] / 2;
    const int* src = edge;        // edge_index[0]
    const int* dst = edge + E;    // edge_index[1]

    // workspace layout: dinv[n f32] | tmp[n*128 f32] | hbuf[n*128 f32]
    //                   | Xb[n*128 bf16] | Wp[128*128 bf16]
    float* ws   = (float*)d_ws;
    float* dinv = ws;
    float* tmp  = dinv + n;
    float* hbuf = tmp + (size_t)n * 128;
    unsigned short* Xb = (unsigned short*)(hbuf + (size_t)n * 128);
    unsigned short* Wp = Xb + (size_t)n * 128;

    // output slices (tuple return order: z, mean, logvar, out)
    float* z_out  = (float*)d_out;
    float* mean_o = z_out  + (size_t)n * L;
    float* lv_o   = mean_o + (size_t)n * L;
    float* out_o  = lv_o   + (size_t)n * L;

    // symmetric normalization: deg includes self loop -> dinv = rsqrt(deg)
    fill_f32<<<ceil_div(n, 256), 256, 0, stream>>>(dinv, 1.0f, n);
    degree_kernel<<<ceil_div(E, 256), 256, 0, stream>>>(dst, dinv, E);
    rsqrt_kernel<<<ceil_div(n, 256), 256, 0, stream>>>(dinv, n);

    // encoder
    run_gcn<128, 128>(feature, W_enc, b_enc, src, dst, dinv, Xb, Wp, tmp, hbuf, n, E, stream);
    run_gcn<128, 64 >(hbuf,   W_mean, b_mean, src, dst, dinv, Xb, Wp, tmp, mean_o, n, E, stream);
    run_gcn<128, 64 >(hbuf,   W_lv,   b_lv,   src, dst, dinv, Xb, Wp, tmp, lv_o,   n, E, stream);

    // reparameterization
    reparam_kernel<<<ceil_div((long)n * L, 256), 256, 0, stream>>>(noise, mean_o, lv_o, z_out, n * L);

    // decoder (hbuf reused as hd)
    run_gcn<64, 128>(z_out, W_d1, b_d1, src, dst, dinv, Xb, Wp, tmp, hbuf, n, E, stream);
    run_gcn<128, 128>(hbuf, W_d2, b_d2, src, dst, dinv, Xb, Wp, tmp, out_o, n, E, stream);
}
